// DeeperGCN_66425964200502
// MI455X (gfx1250) — matrix operand order repaired and checked by reading, not compiled, not run on hardware
//
#include <hip/hip_runtime.h>

#define NN   50000
#define EE   800000
#define FIN  8
#define DD   64
#define D2   128
#define LL   5
#define OUTC 112

typedef __attribute__((ext_vector_type(16))) __bf16 v16bf;
typedef __attribute__((ext_vector_type(8)))  float  v8f;
typedef __attribute__((ext_vector_type(4)))  float  f4;

// ---------------- helpers ----------------

__device__ __forceinline__ void atomic_max_f32(float* addr, float v) {
  // float max via integer ordering trick (works for mixed signs, init = -inf)
  if (v >= 0.0f) atomicMax((int*)addr, __float_as_int(v));
  else           atomicMin((unsigned int*)addr, __float_as_uint(v));
}

__device__ __forceinline__ void atomic_add_f32(float* addr, float v) {
  __hip_atomic_fetch_add(addr, v, __ATOMIC_RELAXED, __HIP_MEMORY_SCOPE_AGENT);
}

__device__ __forceinline__ float wave_sum(float v) {
#pragma unroll
  for (int o = 16; o > 0; o >>= 1) v += __shfl_xor(v, o, 32);
  return v;
}

// ---------------- weight prepack: f32 [K,NCOL] -> bf16 WMMA B-fragment order ------
// Packed layout: [coltile][kt][lane][16 contiguous bf16], so each lane's fragment is
// one 32-byte contiguous load. lane<16 -> K {0..7,16..23} of k-block, lane>=16 -> K {8..15,24..31},
// column = coltile*16 + (lane&15)  (CDNA5 ISA 7.12.2 B layout).

template <int K, int NCOL>
__global__ void gcn_prepack_b(const float* __restrict__ W, __bf16* __restrict__ P) {
  const int total = K * NCOL;
  int idx = blockIdx.x * 256 + threadIdx.x;
  if (idx >= total) return;
  const float* Wl = W + (size_t)blockIdx.y * total;   // layer = blockIdx.y
  __bf16*      Pl = P + (size_t)blockIdx.y * total;
  const int nkt = K / 32;
  int i    = idx & 15;
  int lane = (idx >> 4) & 31;
  int kt   = (idx >> 9) % nkt;
  int ct   = (idx >> 9) / nkt;
  int k = kt * 32 + (lane >> 4) * 8 + (i & 7) + (i >> 3) * 16;
  int n = ct * 16 + (lane & 15);
  Pl[idx] = (__bf16)Wl[k * NCOL + n];
}

// ---------------- node encoder: h = x @ node_W + node_b ----------------

__global__ void gcn_node_encode(const float* __restrict__ x, const float* __restrict__ W,
                                const float* __restrict__ b, float* __restrict__ h) {
  int i = blockIdx.x * 256 + threadIdx.x;     // NN*DD threads exactly
  int node = i >> 6, j = i & 63;
  const float* xr = x + node * FIN;
  float acc = b[j];
#pragma unroll
  for (int k = 0; k < FIN; ++k) acc += xr[k] * W[k * DD + j];
  h[i] = acc;
}

// ---------------- per-layer softmax-aggregation state init ----------------

__global__ void gcn_init_msw(float* __restrict__ m, float* __restrict__ s, float* __restrict__ w) {
  int i = blockIdx.x * 256 + threadIdx.x;     // NN*DD threads exactly
  m[i] = -__builtin_inff();
  s[i] = 0.0f;
  w[i] = 0.0f;
}

// ---------------- edge pass 1: per-dst channel max of logits ----------------
// ea recomputed on the fly from edge_attr (25.6MB/pass) instead of a 205MB buffer;
// h[src] gathers and m/s/w atomics hit L2 (12.8MB arrays vs 192MB L2).

__global__ void gcn_edge_max(const int* __restrict__ src, const int* __restrict__ dst,
                             const float* __restrict__ attr, const float* __restrict__ eW,
                             const float* __restrict__ eb, const float* __restrict__ hin,
                             const float* __restrict__ tarr, int layer,
                             float* __restrict__ mmax) {
  __shared__ float sW[FIN * DD + DD];
  for (int i = threadIdx.x; i < FIN * DD + DD; i += 256)
    sW[i] = (i < FIN * DD) ? eW[i] : eb[i - FIN * DD];
  __syncthreads();

  int gid = blockIdx.x * 256 + threadIdx.x;   // EE*DD threads exactly
  int e = gid >> 6, j = gid & 63;
  const float t = tarr[layer];
  const float* ar = attr + e * FIN;
  float ea = sW[FIN * DD + j];
#pragma unroll
  for (int k = 0; k < FIN; ++k) ea += ar[k] * sW[k * DD + j];
  int sN = src[e], dN = dst[e];
  float msg = fmaxf(hin[sN * DD + j] + ea, 0.0f) + 1e-7f;
  atomic_max_f32(&mmax[dN * DD + j], msg * t);
}

// ---------------- edge pass 2: accumulate sum(e) and sum(msg*e) ----------------

__global__ void gcn_edge_accum(const int* __restrict__ src, const int* __restrict__ dst,
                               const float* __restrict__ attr, const float* __restrict__ eW,
                               const float* __restrict__ eb, const float* __restrict__ hin,
                               const float* __restrict__ tarr, int layer,
                               const float* __restrict__ mmax,
                               float* __restrict__ ssum, float* __restrict__ wsum) {
  __shared__ float sW[FIN * DD + DD];
  for (int i = threadIdx.x; i < FIN * DD + DD; i += 256)
    sW[i] = (i < FIN * DD) ? eW[i] : eb[i - FIN * DD];
  __syncthreads();

  int gid = blockIdx.x * 256 + threadIdx.x;
  int e = gid >> 6, j = gid & 63;
  const float t = tarr[layer];
  const float* ar = attr + e * FIN;
  float ea = sW[FIN * DD + j];
#pragma unroll
  for (int k = 0; k < FIN; ++k) ea += ar[k] * sW[k * DD + j];
  int sN = src[e], dN = dst[e];
  float msg = fmaxf(hin[sN * DD + j] + ea, 0.0f) + 1e-7f;
  float ex = __expf(msg * t - mmax[dN * DD + j]);    // exponent <= 0, no overflow
  atomic_add_f32(&ssum[dN * DD + j], ex);
  atomic_add_f32(&wsum[dN * DD + j], msg * ex);
}

// ---------------- out = wsum/(ssum+1e-16) + hin  (in place into wsum) ----------------

__global__ void gcn_aggregate(const float* __restrict__ ssum, const float* __restrict__ hin,
                              float* __restrict__ wsum) {
  int i = blockIdx.x * 256 + threadIdx.x;     // NN*DD
  wsum[i] = wsum[i] / (ssum[i] + 1e-16f) + hin[i];
}

// ---------------- LayerNorm + ReLU, one wave per row (wave32) ----------------

__global__ void gcn_ln_relu_d64(const float* __restrict__ in, const float* __restrict__ g,
                                const float* __restrict__ b, float* __restrict__ out) {
  int wv = (blockIdx.x * 256 + threadIdx.x) >> 5;   // 50000 waves exactly
  int lane = threadIdx.x & 31;
  const float* r = in + wv * DD;
  float v0 = r[lane], v1 = r[lane + 32];
  float mu = wave_sum(v0 + v1) * (1.0f / DD);
  float d0 = v0 - mu, d1 = v1 - mu;
  float var = wave_sum(d0 * d0 + d1 * d1) * (1.0f / DD);
  float rs = rsqrtf(var + 1e-5f);
  float* o = out + wv * DD;
  o[lane]      = fmaxf(d0 * rs * g[lane]      + b[lane],      0.0f);
  o[lane + 32] = fmaxf(d1 * rs * g[lane + 32] + b[lane + 32], 0.0f);
}

__global__ void gcn_ln_relu_d128(const float* __restrict__ in, const float* __restrict__ g,
                                 const float* __restrict__ b, float* __restrict__ out) {
  int wv = (blockIdx.x * 256 + threadIdx.x) >> 5;
  int lane = threadIdx.x & 31;
  const float* r = in + wv * D2;
  float v0 = r[lane], v1 = r[lane + 32], v2 = r[lane + 64], v3 = r[lane + 96];
  float mu = wave_sum(v0 + v1 + v2 + v3) * (1.0f / D2);
  float d0 = v0 - mu, d1 = v1 - mu, d2 = v2 - mu, d3 = v3 - mu;
  float var = wave_sum(d0 * d0 + d1 * d1 + d2 * d2 + d3 * d3) * (1.0f / D2);
  float rs = rsqrtf(var + 1e-5f);
  float* o = out + wv * D2;
  o[lane]      = fmaxf(d0 * rs * g[lane]      + b[lane],      0.0f);
  o[lane + 32] = fmaxf(d1 * rs * g[lane + 32] + b[lane + 32], 0.0f);
  o[lane + 64] = fmaxf(d2 * rs * g[lane + 64] + b[lane + 64], 0.0f);
  o[lane + 96] = fmaxf(d3 * rs * g[lane + 96] + b[lane + 96], 0.0f);
}

// ---------------- WMMA bf16 GEMM: Cout = A@B + bias (+ Cin residual) ----------------
// One wave per 16x16 output tile; blockIdx = row tile (50000 = 3125*16 exactly),
// wave id = column tile -> EXEC all-ones at every WMMA (ISA requirement).
// A: fp32 row-major, loaded as float4 (16B-aligned) and cvt'd to bf16 on the fly.
// B: bf16 prepacked in fragment order -> one contiguous 32B load per lane per k-step.
// RES is compile-time so the epilogue is branch-free straight-line code.

template <int K, int NCOL, bool RES>
__global__ void gcn_wmma_gemm(const float* __restrict__ A, const __bf16* __restrict__ B,
                              const float* __restrict__ bias, const float* __restrict__ Cin,
                              float* __restrict__ Cout) {
  const int lane = threadIdx.x & 31;
  const int wave = threadIdx.x >> 5;
  const int row  = blockIdx.x * 16;
  const int r    = row + (lane & 15);
  const int n    = wave * 16 + (lane & 15);
  const int hi   = (lane & 16) ? 1 : 0;

  v8f c = {0.f, 0.f, 0.f, 0.f, 0.f, 0.f, 0.f, 0.f};
#pragma unroll
  for (int kt = 0; kt < K / 32; ++kt) {
    // A fragment: lanes 0-15 hold K {0..7,16..23}, lanes 16-31 hold K {8..15,24..31}
    const float* ap = A + r * K + kt * 32 + hi * 8;
    f4 a0 = *(const f4*)(ap);
    f4 a1 = *(const f4*)(ap + 4);
    f4 a2 = *(const f4*)(ap + 16);
    f4 a3 = *(const f4*)(ap + 20);
    v16bf a, b;
#pragma unroll
    for (int i = 0; i < 4; ++i) {
      a[i]      = (__bf16)a0[i];
      a[4 + i]  = (__bf16)a1[i];
      a[8 + i]  = (__bf16)a2[i];
      a[12 + i] = (__bf16)a3[i];
    }
    // B fragment: prepacked contiguous 16 bf16 per lane
    b = *(const v16bf*)(B + ((size_t)wave * (K / 32) + kt) * 512 + lane * 16);
    c = __builtin_amdgcn_wmma_f32_16x16x32_bf16(false, a, false, b, (short)0, c, false, false);
  }

  // C/D layout: VGPR i -> row (row + hi*8 + i), col n
  const int m0 = row + hi * 8;
  const float bn = bias[n];
#pragma unroll
  for (int i = 0; i < 8; ++i) {
    float v = c[i] + bn;
    if (RES) v += Cin[(m0 + i) * NCOL + n];
    Cout[(m0 + i) * NCOL + n] = v;
  }
}

// ---------------- launcher ----------------

extern "C" void kernel_launch(void* const* d_in, const int* in_sizes, int n_in,
                              void* d_out, int out_size, void* d_ws, size_t ws_size,
                              hipStream_t stream) {
  (void)in_sizes; (void)n_in; (void)out_size; (void)ws_size;

  const float* x     = (const float*)d_in[0];
  const float* eattr = (const float*)d_in[1];
  const int*   eidx  = (const int*)d_in[2];
  const int*   src   = eidx;
  const int*   dst   = eidx + EE;
  const float* nodeW = (const float*)d_in[3];
  const float* nodeB = (const float*)d_in[4];
  const float* edgeW = (const float*)d_in[5];
  const float* edgeB = (const float*)d_in[6];
  const float* t     = (const float*)d_in[7];
  const float* W1    = (const float*)d_in[8];
  const float* b1    = (const float*)d_in[9];
  const float* mlng  = (const float*)d_in[10];
  const float* mlnb  = (const float*)d_in[11];
  const float* W2    = (const float*)d_in[12];
  const float* b2    = (const float*)d_in[13];
  const float* lng   = (const float*)d_in[14];
  const float* lnb   = (const float*)d_in[15];
  const float* linW  = (const float*)d_in[16];
  const float* linB  = (const float*)d_in[17];

  // workspace layout (~90 MB); all offsets 32B-aligned
  char* p = (char*)d_ws;
  float* h   = (float*)p; p += (size_t)NN * DD * 4;
  float* hin = (float*)p; p += (size_t)NN * DD * 4;
  float* m   = (float*)p; p += (size_t)NN * DD * 4;
  float* s   = (float*)p; p += (size_t)NN * DD * 4;
  float* w   = (float*)p; p += (size_t)NN * DD * 4;
  float* z1  = (float*)p; p += (size_t)NN * D2 * 4;
  __bf16* W1b = (__bf16*)p; p += (size_t)LL * DD * D2 * 2;
  __bf16* W2b = (__bf16*)p; p += (size_t)LL * D2 * DD * 2;
  __bf16* LWb = (__bf16*)p; p += (size_t)DD * OUTC * 2;

  const int ND_BLOCKS = NN * DD / 256;   // 12500
  const int ED_BLOCKS = EE * DD / 256;   // 200000
  const int LN_BLOCKS = NN / 8;          // 6250 (8 waves/block, 1 wave/row)
  const int RT_BLOCKS = NN / 16;         // 3125 row tiles

  gcn_prepack_b<DD, D2><<<dim3(32, LL), 256, 0, stream>>>(W1, W1b);     // 8192/layer
  gcn_prepack_b<D2, DD><<<dim3(32, LL), 256, 0, stream>>>(W2, W2b);     // 8192/layer
  gcn_prepack_b<DD, OUTC><<<dim3(28, 1), 256, 0, stream>>>(linW, LWb);  // 7168
  gcn_node_encode<<<ND_BLOCKS, 256, 0, stream>>>(x, nodeW, nodeB, h);

  for (int l = 0; l < LL; ++l) {
    const float* hl;
    if (l == 0) {
      hl = h;  // layer 0: conv on raw encoder output
    } else {
      gcn_ln_relu_d64<<<LN_BLOCKS, 256, 0, stream>>>(h, lng + l * DD, lnb + l * DD, hin);
      hl = hin;
    }
    gcn_init_msw<<<ND_BLOCKS, 256, 0, stream>>>(m, s, w);
    gcn_edge_max<<<ED_BLOCKS, 256, 0, stream>>>(src, dst, eattr, edgeW, edgeB, hl, t, l, m);
    gcn_edge_accum<<<ED_BLOCKS, 256, 0, stream>>>(src, dst, eattr, edgeW, edgeB, hl, t, l, m, s, w);
    gcn_aggregate<<<ND_BLOCKS, 256, 0, stream>>>(s, hl, w);   // w := out (incl. root skip)

    // MLP: z1 = relu(LN(out @ W1 + b1)); h = z1 @ W2 + b2 (+ h residual for l>0)
    gcn_wmma_gemm<DD, D2, false><<<RT_BLOCKS, 256, 0, stream>>>(
        w, W1b + (size_t)l * DD * D2, b1 + l * D2, h, z1);
    gcn_ln_relu_d128<<<LN_BLOCKS, 256, 0, stream>>>(z1, mlng + l * D2, mlnb + l * D2, z1);
    if (l == 0)
      gcn_wmma_gemm<D2, DD, false><<<RT_BLOCKS, 128, 0, stream>>>(
          z1, W2b + (size_t)l * D2 * DD, b2 + l * DD, h, h);
    else
      gcn_wmma_gemm<D2, DD, true><<<RT_BLOCKS, 128, 0, stream>>>(
          z1, W2b + (size_t)l * D2 * DD, b2 + l * DD, h, h);
  }

  // final: out = relu(LN(h; layer-0 norm)) @ lin_W + lin_b
  gcn_ln_relu_d64<<<LN_BLOCKS, 256, 0, stream>>>(h, lng, lnb, hin);
  gcn_wmma_gemm<DD, OUTC, false><<<RT_BLOCKS, 224, 0, stream>>>(hin, LWb, linB, h,
                                                                (float*)d_out);
}
